// RippleAttention_62749472194587
// MI455X (gfx1250) — matrix operand order repaired and checked by reading, not compiled, not run on hardware
//
#include <hip/hip_runtime.h>
#include <hip/hip_bf16.h>
#include <math.h>
#include <stdint.h>

typedef __attribute__((ext_vector_type(16))) _Float16 v16h;
typedef __attribute__((ext_vector_type(8)))  float    v8f;

__device__ __forceinline__ v8f wmma_f16(v16h a, v16h b, v8f c) {
  return __builtin_amdgcn_wmma_f32_16x16x32_f16(false, a, false, b, (short)0, c, false, false);
}

// Low 32 bits of a generic pointer to LDS == LDS byte offset (ISA 10.2 aperture map)
__device__ __forceinline__ uint32_t lds_off32(const void* p) {
  return (uint32_t)(uintptr_t)p;
}
// CDNA5 async copy: 16B per active lane, SGPR base + per-lane 32-bit byte offset (GVS)
__device__ __forceinline__ void async_b128(uint32_t ldsb, uint32_t voffb, const void* sbase) {
  asm volatile("global_load_async_to_lds_b128 %0, %1, %2"
               :: "v"(ldsb), "v"(voffb), "s"(sbase) : "memory");
}
template<int N>
__device__ __forceinline__ void wait_async() {
  asm volatile("s_wait_asynccnt %0" :: "i"(N) : "memory");
}

// ---- fragment loaders per CDNA5 ISA 7.12.2 (wave32) ----
template<int STR>
__device__ __forceinline__ v16h frag_a(const _Float16 (&A)[64][STR], int mt, int ks, int lane) {
  v16h a;
  const int m  = mt * 16 + (lane & 15);
  const int kb = ((lane < 16) ? 0 : 8) + ks;
#pragma unroll
  for (int i = 0; i < 4; ++i) {
    a[2*i]     = A[m][kb + 2*i];
    a[2*i + 1] = A[m][kb + 2*i + 1];
    a[2*i + 8] = A[m][kb + 16 + 2*i];
    a[2*i + 9] = A[m][kb + 16 + 2*i + 1];
  }
  return a;
}
// B stored transposed in LDS: Bt[n][k] -> lane fragment = two runs of 8 halves (table k order)
template<int STR>
__device__ __forceinline__ v16h frag_bT(const _Float16 (&Bt)[64][STR], int nt, int ks, int lane) {
  v16h b;
  const int n  = nt * 16 + (lane & 15);
  const int kb = ((lane < 16) ? 0 : 8) + ks;
#pragma unroll
  for (int j = 0; j < 16; ++j) b[j] = Bt[n][kb + j + ((j >= 8) ? 8 : 0)];
  return b;
}
template<int STR>
__device__ __forceinline__ void store_c(float (&C)[64][STR], const v8f& c, int mt, int nt, int lane) {
  const int n  = nt * 16 + (lane & 15);
  const int mb = mt * 16 + ((lane < 16) ? 0 : 8);
#pragma unroll
  for (int r = 0; r < 8; ++r) C[mb + r][n] = c[r];
}

// ============ Prep 1: x (f32) -> xh (f16) ============
__global__ __launch_bounds__(256)
void k_cvt_x(const float* __restrict__ x, _Float16* __restrict__ xh) {
  int i = (blockIdx.x * 256 + threadIdx.x) * 8;
  float4 a = *(const float4*)(x + i);
  float4 b = *(const float4*)(x + i + 4);
  _Float16 h8[8] = {(_Float16)a.x, (_Float16)a.y, (_Float16)a.z, (_Float16)a.w,
                    (_Float16)b.x, (_Float16)b.y, (_Float16)b.z, (_Float16)b.w};
  *(int4*)(xh + i) = *(const int4*)h8;
}

// ============ Prep 2: big W (768x768 f32) -> transposed f16 ============
__global__ __launch_bounds__(256)
void k_cvt_wT(const float* __restrict__ Wq, const float* __restrict__ Wk,
              const float* __restrict__ Wv, const float* __restrict__ Wo,
              _Float16* __restrict__ qT, _Float16* __restrict__ kT,
              _Float16* __restrict__ vT, _Float16* __restrict__ oT) {
  const int k = blockIdx.x, z = blockIdx.y;
  const float* W = (z == 0) ? Wq : (z == 1) ? Wk : (z == 2) ? Wv : Wo;
  _Float16*    T = (z == 0) ? qT : (z == 1) ? kT : (z == 2) ? vT : oT;
#pragma unroll
  for (int it = 0; it < 3; ++it) {
    int n = threadIdx.x + it * 256;
    T[(size_t)n * 768 + k] = (_Float16)W[(size_t)k * 768 + n];
  }
}

// ============ Prep 3: small 64x64 weights (pool x3, exp x3) -> transposed f16 ============
__global__ __launch_bounds__(256)
void k_cvt_small(const float* __restrict__ pW, const float* __restrict__ eW,
                 _Float16* __restrict__ swT) {
  const int z = blockIdx.x;                                // 0..5
  const float* src = (z < 3) ? (pW + z * 4096) : (eW + (z - 3) * 4096);
  _Float16*    dst = swT + z * 4096;
#pragma unroll
  for (int it = 0; it < 16; ++it) {
    int lin = threadIdx.x + it * 256;
    int k = lin >> 6, n = lin & 63;
    dst[n * 64 + k] = (_Float16)src[k * 64 + n];
  }
}

// ============ Kernel 1: fused QKV projection + elu+1 (double-buffered async) ============
__global__ __launch_bounds__(256)
void k_qkv(const _Float16* __restrict__ xh,
           const _Float16* __restrict__ WqT, const _Float16* __restrict__ WkT,
           const _Float16* __restrict__ WvT,
           _Float16* __restrict__ qb, _Float16* __restrict__ kb,
           _Float16* __restrict__ vb) {
  __shared__ _Float16 Alds[2][64][40];
  __shared__ _Float16 Blds[3][2][64][40];
  __shared__ float    Clds[64][68];
  const int t = threadIdx.x, lane = t & 31, wave = t >> 5;
  const int m0 = blockIdx.x * 64, c0 = blockIdx.y * 64;
  const int mt = wave >> 1, nt0 = (wave & 1) * 2;
  const int ar = t >> 2, ac = (t & 3) * 8;       // one 16B chunk per lane
  uint32_t aL[2], bL[3][2];
#pragma unroll
  for (int bf = 0; bf < 2; ++bf) {
    aL[bf] = lds_off32(&Alds[bf][ar][ac]);
#pragma unroll
    for (int z = 0; z < 3; ++z) bL[z][bf] = lds_off32(&Blds[z][bf][ar][ac]);
  }
  const uint32_t aOff = (uint32_t)(((m0 + ar) * 768 + ac) * 2);
  const uint32_t bOff = (uint32_t)(((c0 + ar) * 768 + ac) * 2);
  auto issue = [&](int tile, int bf) {
    uint32_t ko = (uint32_t)(tile * 64);         // 32 halves = 64 bytes along k
    async_b128(aL[bf],    aOff + ko, xh);
    async_b128(bL[0][bf], bOff + ko, WqT);
    async_b128(bL[1][bf], bOff + ko, WkT);
    async_b128(bL[2][bf], bOff + ko, WvT);
  };
  v8f acc[3][2] = {};
  issue(0, 0);
  for (int tile = 0; tile < 24; ++tile) {
    const int bf = tile & 1;
    if (tile + 1 < 24) { issue(tile + 1, bf ^ 1); wait_async<4>(); }
    else               { wait_async<0>(); }
    __syncthreads();
    v16h a = frag_a(Alds[bf], mt, 0, lane);
#pragma unroll
    for (int z = 0; z < 3; ++z) {
      v16h b0 = frag_bT(Blds[z][bf], nt0, 0, lane);
      v16h b1 = frag_bT(Blds[z][bf], nt0 + 1, 0, lane);
      acc[z][0] = wmma_f16(a, b0, acc[z][0]);
      acc[z][1] = wmma_f16(a, b1, acc[z][1]);
    }
    __syncthreads();
  }
  const int hh = c0 >> 6;                        // block covers exactly one head
#pragma unroll
  for (int z = 0; z < 3; ++z) {
    store_c(Clds, acc[z][0], mt, nt0, lane);
    store_c(Clds, acc[z][1], mt, nt0 + 1, lane);
    __syncthreads();
    _Float16* dst = (z == 0) ? qb : (z == 1) ? kb : vb;
#pragma unroll
    for (int it = 0; it < 2; ++it) {
      int i = t + it * 256;
      int r = i >> 3, c8 = (i & 7) * 8;
      int gr = m0 + r, bbi = gr >> 12, n = gr & 4095;
      _Float16 h8[8];
#pragma unroll
      for (int u = 0; u < 8; ++u) {
        float vv = Clds[r][c8 + u];
        if (z < 2) vv = (vv > 0.f) ? (vv + 1.f) : __expf(vv);   // elu(x)+1
        h8[u] = (_Float16)vv;
      }
      *(int4*)(dst + (((size_t)(bbi * 12 + hh) * 4096) + n) * 64 + c8) = *(const int4*)h8;
    }
    __syncthreads();
  }
}

// ============ Kernel 2: pooling stage (k: pool+MLP+LN+GELU, v: pool) ============
__global__ __launch_bounds__(256)
void k_pool(const _Float16* __restrict__ kin, const _Float16* __restrict__ vin,
            const _Float16* __restrict__ WT, const float* __restrict__ bias,
            const float* __restrict__ g, const float* __restrict__ be,
            _Float16* __restrict__ kout, _Float16* __restrict__ vout, int s_in) {
  __shared__ _Float16 Alds[64][72];
  __shared__ _Float16 Blds[64][72];     // Bt[n][k]
  __shared__ float    Clds[64][68];
  const int t = threadIdx.x, lane = t & 31, wave = t >> 5;
  const int bh = blockIdx.x, r0 = blockIdx.y * 64;
  const int s_out = s_in >> 1;
  const size_t inbase  = (size_t)bh * s_in * 64;
  const size_t outbase = (size_t)bh * s_out * 64;
  union H8 { int4 q; _Float16 h[8]; };
#pragma unroll
  for (int it = 0; it < 2; ++it) {
    int i = t + it * 256;
    int r = i >> 3, c8 = (i & 7) * 8;
    int ir = (r0 + r) * 2;
    H8 a0, a1, b0, b1, oa, ov;
    a0.q = *(const int4*)(kin + inbase + (size_t)ir * 64 + c8);
    a1.q = *(const int4*)(kin + inbase + (size_t)(ir + 1) * 64 + c8);
    b0.q = *(const int4*)(vin + inbase + (size_t)ir * 64 + c8);
    b1.q = *(const int4*)(vin + inbase + (size_t)(ir + 1) * 64 + c8);
#pragma unroll
    for (int u = 0; u < 8; ++u) {
      oa.h[u] = (_Float16)(0.5f * ((float)a0.h[u] + (float)a1.h[u]));
      ov.h[u] = (_Float16)(0.5f * ((float)b0.h[u] + (float)b1.h[u]));
    }
    *(int4*)&Alds[r][c8] = oa.q;
    *(int4*)(vout + outbase + (size_t)(r0 + r) * 64 + c8) = ov.q;
    *(int4*)&Blds[r][c8] = *(const int4*)(WT + r * 64 + c8);
  }
  __syncthreads();
  const int mt = wave >> 1, nt0 = (wave & 1) * 2;
  v8f acc0 = {}, acc1 = {};
#pragma unroll
  for (int ks = 0; ks < 64; ks += 32) {
    v16h a  = frag_a (Alds, mt, ks, lane);
    v16h b0 = frag_bT(Blds, nt0, ks, lane);
    v16h b1 = frag_bT(Blds, nt0 + 1, ks, lane);
    acc0 = wmma_f16(a, b0, acc0);
    acc1 = wmma_f16(a, b1, acc1);
  }
  store_c(Clds, acc0, mt, nt0, lane);
  store_c(Clds, acc1, mt, nt0 + 1, lane);
  __syncthreads();
  if (t < 64) {
    const int r = t;
    float mu = 0.f, s2 = 0.f;
#pragma unroll
    for (int c = 0; c < 64; ++c) {
      float y = Clds[r][c] + bias[c];
      Clds[r][c] = y;
      mu += y; s2 += y * y;
    }
    mu *= (1.f / 64.f);
    s2 = s2 * (1.f / 64.f) - mu * mu;
    float inv = rsqrtf(s2 + 1e-5f);
    for (int c8 = 0; c8 < 64; c8 += 8) {
      _Float16 h8[8];
#pragma unroll
      for (int u = 0; u < 8; ++u) {
        float uu = (Clds[r][c8 + u] - mu) * inv * g[c8 + u] + be[c8 + u];
        h8[u] = (_Float16)(0.5f * uu * (1.f + erff(uu * 0.70710678118654752f)));
      }
      *(int4*)(kout + outbase + (size_t)(r0 + r) * 64 + c8) = *(const int4*)h8;
    }
  }
}

// ============ Kernel 3: kvT[e][d] = (kc^T vc)^T / sqrt(s), ksum = sum(kc)/sqrt(s) ============
__global__ __launch_bounds__(256)
void k_kv(const _Float16* __restrict__ kc, const _Float16* __restrict__ vc,
          _Float16* __restrict__ kvT, float* __restrict__ ksum, int s) {
  __shared__ _Float16 Alds[64][40];     // A[d][n]  (= kc^T)
  __shared__ _Float16 Blds[64][40];     // Bt[e][n] (= vc^T)
  __shared__ float    Clds[64][68];
  const int t = threadIdx.x, lane = t & 31, wave = t >> 5;
  const int bh = blockIdx.x;
  const size_t base = (size_t)bh * s * 64;
  const int mt = wave >> 1, nt0 = (wave & 1) * 2;
  v8f acc0 = {}, acc1 = {};
  for (int n0 = 0; n0 < s; n0 += 32) {
#pragma unroll
    for (int it = 0; it < 8; ++it) {
      int lin = t + it * 256;
      int r = lin >> 6, c = lin & 63;
      Alds[c][r] = kc[base + (size_t)(n0 + r) * 64 + c];
      Blds[c][r] = vc[base + (size_t)(n0 + r) * 64 + c];
    }
    __syncthreads();
    v16h a  = frag_a (Alds, mt, 0, lane);
    v16h b0 = frag_bT(Blds, nt0, 0, lane);
    v16h b1 = frag_bT(Blds, nt0 + 1, 0, lane);
    acc0 = wmma_f16(a, b0, acc0);
    acc1 = wmma_f16(a, b1, acc1);
    __syncthreads();
  }
  store_c(Clds, acc0, mt, nt0, lane);
  store_c(Clds, acc1, mt, nt0 + 1, lane);
  __syncthreads();
  const float sc = rsqrtf((float)s);
#pragma unroll
  for (int it = 0; it < 2; ++it) {      // store transposed for the consumer
    int i = t + it * 256;
    int e = i >> 3, d8 = (i & 7) * 8;
    _Float16 h8[8];
#pragma unroll
    for (int u = 0; u < 8; ++u) h8[u] = (_Float16)(Clds[d8 + u][e] * sc);
    *(int4*)(kvT + (size_t)bh * 4096 + e * 64 + d8) = *(const int4*)h8;
  }
  if (t < 64) {
    float acc = 0.f;
    for (int n = 0; n < s; ++n) acc += (float)kc[base + (size_t)n * 64 + t];
    ksum[bh * 64 + t] = acc * sc;
  }
}

// ==== Kernel 4: fused  h = expand0(expand1(expand2( (q@kv)/(q.ksum+1e-6) )))  ====
__global__ __launch_bounds__(256)
void k_apply_expand(const _Float16* __restrict__ q, const _Float16* __restrict__ kvT,
                    const float* __restrict__ ksum, const _Float16* __restrict__ eWT,
                    const float* __restrict__ eb, const float* __restrict__ eg,
                    const float* __restrict__ ebe, _Float16* __restrict__ hout) {
  __shared__ _Float16 Alds[64][72];
  __shared__ _Float16 Bt[4][64][72];    // [0]=kvT, [1..3]=exp stage 2,1,0
  __shared__ float    Clds[64][68];
  const int t = threadIdx.x, lane = t & 31, wave = t >> 5;
  const int bh = blockIdx.x, r0 = blockIdx.y * 64;
  const int mt = wave >> 1, nt0 = (wave & 1) * 2;
  const size_t qbase = ((size_t)bh * 4096 + r0) * 64;
#pragma unroll
  for (int it = 0; it < 2; ++it) {      // A = q rows
    int i = t + it * 256;
    int r = i >> 3, c8 = (i & 7) * 8;
    *(int4*)&Alds[r][c8] = *(const int4*)(q + qbase + (size_t)r * 64 + c8);
  }
#pragma unroll
  for (int it = 0; it < 8; ++it) {      // 4 transposed B matrices, contiguous copies
    int i = t + it * 256;               // 0..2047 chunks of 8 halves
    int j = i >> 9, r = (i >> 3) & 63, c8 = (i & 7) * 8;
    const _Float16* src = (j == 0) ? (kvT + (size_t)bh * 4096)
                                   : (eWT + (size_t)(3 - j) * 4096);
    *(int4*)&Bt[j][r][c8] = *(const int4*)(src + r * 64 + c8);
  }
  __syncthreads();
  auto gemm = [&](const _Float16 (&Bm)[64][72]) {
    v8f a0 = {}, a1 = {};
#pragma unroll
    for (int ks = 0; ks < 64; ks += 32) {
      v16h a  = frag_a (Alds, mt, ks, lane);
      v16h b0 = frag_bT(Bm, nt0, ks, lane);
      v16h b1 = frag_bT(Bm, nt0 + 1, ks, lane);
      a0 = wmma_f16(a, b0, a0);
      a1 = wmma_f16(a, b1, a1);
    }
    store_c(Clds, a0, mt, nt0, lane);
    store_c(Clds, a1, mt, nt0 + 1, lane);
  };
  // --- attention apply + normalize ---
  gemm(Bt[0]);
  __syncthreads();
  if (t < 64) {
    const int r = t;
    float den = 1e-6f;
#pragma unroll
    for (int c = 0; c < 64; ++c) den += (float)Alds[r][c] * ksum[bh * 64 + c];
    float inv = 1.f / den;
#pragma unroll
    for (int c = 0; c < 64; ++c) Alds[r][c] = (_Float16)(Clds[r][c] * inv);
  }
  __syncthreads();
  // --- three expand MLP stages (reversed order: 2,1,0) ---
#pragma unroll
  for (int j = 1; j <= 3; ++j) {
    gemm(Bt[j]);
    __syncthreads();
    const int st = 3 - j;
    if (t < 64) {
      const int r = t;
      float mu = 0.f, s2 = 0.f;
#pragma unroll
      for (int c = 0; c < 64; ++c) {
        float y = Clds[r][c] + eb[st * 64 + c];
        Clds[r][c] = y;
        mu += y; s2 += y * y;
      }
      mu *= (1.f / 64.f);
      s2 = s2 * (1.f / 64.f) - mu * mu;
      float inv = rsqrtf(s2 + 1e-5f);
#pragma unroll
      for (int c = 0; c < 64; ++c) {
        float uu = (Clds[r][c] - mu) * inv * eg[st * 64 + c] + ebe[st * 64 + c];
        Alds[r][c] = (_Float16)(0.5f * uu * (1.f + erff(uu * 0.70710678118654752f)));
      }
    }
    __syncthreads();
  }
#pragma unroll
  for (int it = 0; it < 2; ++it) {      // final store
    int i = t + it * 256;
    int r = i >> 3, c8 = (i & 7) * 8;
    *(int4*)(hout + qbase + (size_t)r * 64 + c8) = *(const int4*)&Alds[r][c8];
  }
}

// ============ Kernel 5: merge heads + output projection (double-buffered async) ============
__global__ __launch_bounds__(256)
void k_out(const _Float16* __restrict__ h, const _Float16* __restrict__ WoT,
           const float* __restrict__ bo, float* __restrict__ out) {
  __shared__ _Float16 Alds[2][64][40];
  __shared__ _Float16 Blds[2][64][40];
  __shared__ float    Clds[64][68];
  const int t = threadIdx.x, lane = t & 31, wave = t >> 5;
  const int m0 = blockIdx.x * 64, c0 = blockIdx.y * 64;
  const int mt = wave >> 1, nt0 = (wave & 1) * 2;
  const int ar = t >> 2, ac = (t & 3) * 8;
  uint32_t aL[2] = {lds_off32(&Alds[0][ar][ac]), lds_off32(&Alds[1][ar][ac])};
  uint32_t bL[2] = {lds_off32(&Blds[0][ar][ac]), lds_off32(&Blds[1][ar][ac])};
  const int gr = m0 + ar, bbi = gr >> 12, n = gr & 4095;
  auto issue = [&](int tile, int bf) {
    int kk = tile * 32;
    int gc = kk + ac, hh = gc >> 6, d = gc & 63;     // 8-chunk stays in one head
    async_b128(aL[bf], (uint32_t)(((((bbi * 12 + hh) * 4096) + n) * 64 + d) * 2), h);
    async_b128(bL[bf], (uint32_t)(((c0 + ar) * 768 + kk + ac) * 2), WoT);
  };
  v8f acc0 = {}, acc1 = {};
  issue(0, 0);
  for (int tile = 0; tile < 24; ++tile) {
    const int bf = tile & 1;
    if (tile + 1 < 24) { issue(tile + 1, bf ^ 1); wait_async<2>(); }
    else               { wait_async<0>(); }
    __syncthreads();
    v16h a  = frag_a (Alds[bf], mt, 0, lane);
    v16h b0 = frag_bT(Blds[bf], nt0, 0, lane);
    v16h b1 = frag_bT(Blds[bf], nt0 + 1, 0, lane);
    acc0 = wmma_f16(a, b0, acc0);
    acc1 = wmma_f16(a, b1, acc1);
    __syncthreads();
  }
  store_c(Clds, acc0, mt, nt0, lane);
  store_c(Clds, acc1, mt, nt0 + 1, lane);
  __syncthreads();
#pragma unroll
  for (int it = 0; it < 4; ++it) {
    int i = t + it * 256;
    int r = i >> 4, c4 = (i & 15) * 4;
    float4 o;
    o.x = Clds[r][c4 + 0] + bo[c0 + c4 + 0];
    o.y = Clds[r][c4 + 1] + bo[c0 + c4 + 1];
    o.z = Clds[r][c4 + 2] + bo[c0 + c4 + 2];
    o.w = Clds[r][c4 + 3] + bo[c0 + c4 + 3];
    *(float4*)(out + (size_t)(m0 + r) * 768 + c0 + c4) = o;
  }
}

extern "C" void kernel_launch(void* const* d_in, const int* in_sizes, int n_in,
                              void* d_out, int out_size, void* d_ws, size_t ws_size,
                              hipStream_t stream) {
  (void)in_sizes; (void)n_in; (void)out_size; (void)ws_size;
  const float* x   = (const float*)d_in[0];
  const float* Wq  = (const float*)d_in[1];
  const float* Wk  = (const float*)d_in[2];
  const float* Wv  = (const float*)d_in[3];
  const float* pW  = (const float*)d_in[4];
  const float* pb  = (const float*)d_in[5];
  const float* pg  = (const float*)d_in[6];
  const float* pbe = (const float*)d_in[7];
  const float* eW  = (const float*)d_in[8];
  const float* eb  = (const float*)d_in[9];
  const float* eg  = (const float*)d_in[10];
  const float* ebe = (const float*)d_in[11];
  const float* Wo  = (const float*)d_in[12];
  const float* bo  = (const float*)d_in[13];
  float* out = (float*)d_out;

  char* w = (char*)d_ws;
  auto carve = [&](size_t bytes) {
    char* p = w;
    w += (bytes + 255) & ~(size_t)255;
    return p;
  };
  const size_t PH = (size_t)96 * 4096 * 64;            // per-head f16 elements
  _Float16* xh   = (_Float16*)carve((size_t)32768 * 768 * 2);
  _Float16* WqT  = (_Float16*)carve((size_t)768 * 768 * 2);
  _Float16* WkT  = (_Float16*)carve((size_t)768 * 768 * 2);
  _Float16* WvT  = (_Float16*)carve((size_t)768 * 768 * 2);
  _Float16* WoT  = (_Float16*)carve((size_t)768 * 768 * 2);
  _Float16* swT  = (_Float16*)carve((size_t)6 * 4096 * 2);
  _Float16* qb   = (_Float16*)carve(PH * 2);
  _Float16* kb   = (_Float16*)carve(PH * 2);
  _Float16* vb   = (_Float16*)carve(PH * 2);
  _Float16* kcA  = (_Float16*)carve((size_t)96 * 2048 * 64 * 2);
  _Float16* vcA  = (_Float16*)carve((size_t)96 * 2048 * 64 * 2);
  _Float16* kcB  = (_Float16*)carve((size_t)96 * 1024 * 64 * 2);
  _Float16* vcB  = (_Float16*)carve((size_t)96 * 1024 * 64 * 2);
  _Float16* kcC  = (_Float16*)carve((size_t)96 * 512 * 64 * 2);
  _Float16* vcC  = (_Float16*)carve((size_t)96 * 512 * 64 * 2);
  _Float16* kvT  = (_Float16*)carve((size_t)96 * 64 * 64 * 2);
  float*    ksmb = (float*)carve((size_t)96 * 64 * 4);
  _Float16* hb   = (_Float16*)carve(PH * 2);

  // 0) precision/layout prep
  k_cvt_x    <<<dim3(12288),  256, 0, stream>>>(x, xh);
  k_cvt_wT   <<<dim3(768, 4), 256, 0, stream>>>(Wq, Wk, Wv, Wo, WqT, WkT, WvT, WoT);
  k_cvt_small<<<dim3(6),      256, 0, stream>>>(pW, eW, swT);
  // 1) fused QKV projection + feature map
  k_qkv<<<dim3(512, 12), 256, 0, stream>>>(xh, WqT, WkT, WvT, qb, kb, vb);
  // 2) three pooling stages
  k_pool<<<dim3(96, 32), 256, 0, stream>>>(kb,  vb,  swT,        pb,       pg,       pbe,       kcA, vcA, 4096);
  k_pool<<<dim3(96, 16), 256, 0, stream>>>(kcA, vcA, swT + 4096, pb + 64,  pg + 64,  pbe + 64,  kcB, vcB, 2048);
  k_pool<<<dim3(96,  8), 256, 0, stream>>>(kcB, vcB, swT + 8192, pb + 128, pg + 128, pbe + 128, kcC, vcC, 1024);
  // 3) kv state + k-sum
  k_kv<<<dim3(96), 256, 0, stream>>>(kcC, vcC, kvT, ksmb, 512);
  // 4) fused attention-apply + 3 expand MLPs (all in LDS)
  k_apply_expand<<<dim3(96, 64), 256, 0, stream>>>(qb, kvT, ksmb, swT + 3 * 4096,
                                                   eb, eg, ebe, hb);
  // 5) merge heads + output projection
  k_out<<<dim3(512, 12), 256, 0, stream>>>(hb, WoT, bo, out);
}
